// Block_88441966559402
// MI455X (gfx1250) — matrix operand order repaired and checked by reading, not compiled, not run on hardware
//
#include <hip/hip_runtime.h>
#include <math.h>

// ---------------------------------------------------------------------------
// Types for CDNA5 WMMA (wave32): D(16x16 f32) = A(16x32 bf16) x B(32x16 bf16) + C
// ---------------------------------------------------------------------------
typedef __bf16 bf16;
typedef __bf16 bf16x8 __attribute__((ext_vector_type(8)));
typedef __bf16 v16bf  __attribute__((ext_vector_type(16)));
typedef float  v8f    __attribute__((ext_vector_type(8)));

// gfx1250 async global->LDS path (guarded; fallback = load + ds_store)
#if defined(__has_builtin)
#if __has_builtin(__builtin_amdgcn_global_load_async_to_lds_b128) && \
    __has_builtin(__builtin_amdgcn_s_wait_asynccnt)
#define USE_ASYNC_LDS 1
#endif
#endif

// Exact pointee type expected by the async builtins: generic vector of 4 ints,
// in AS1 (global) for the source and AS3 (LDS) for the destination.
typedef int i32x4 __attribute__((vector_size(16)));
typedef __attribute__((address_space(1))) i32x4 g_i32x4;
typedef __attribute__((address_space(3))) i32x4 l_i32x4;

__device__ inline v16bf pack2(bf16x8 lo, bf16x8 hi) {
  v16bf r;
#pragma unroll
  for (int e = 0; e < 8; ++e) { r[e] = lo[e]; r[e + 8] = hi[e]; }
  return r;
}

__device__ inline v8f wmma_bf16(v16bf a, v16bf b, v8f c) {
  return __builtin_amdgcn_wmma_f32_16x16x32_bf16(
      /*neg_a=*/false, a, /*neg_b=*/false, b,
      /*c_mod=*/(short)0, c, /*reuse_a=*/false, /*reuse_b=*/false);
}

__device__ inline float gelu_exact(float x) {
  return 0.5f * x * (1.0f + erff(x * 0.70710678118654752f));
}

// Problem constants
#define DD   768
#define DOWN 192
#define HID  3072
#define TT   8
#define NN   196
#define HEADS 12
#define HD   64
#define MT   6272   // 4*8*196 t tokens
#define MS   6304   // 32*197 s tokens

// ---------------------------------------------------------------------------
// f32 -> bf16 convert (weights)
// ---------------------------------------------------------------------------
__global__ void f2b_kernel(const float* __restrict__ in, bf16* __restrict__ out, int n) {
  int i = blockIdx.x * 256 + threadIdx.x;
  if (i < n) out[i] = (bf16)in[i];
}

__global__ void qkv_bias_kernel(const float* __restrict__ qb, const float* __restrict__ vb,
                                float* __restrict__ out) {
  int i = blockIdx.x * 256 + threadIdx.x;
  if (i < 3 * DD) out[i] = (i < DD) ? qb[i] : ((i < 2 * DD) ? 0.0f : vb[i - 2 * DD]);
}

// ---------------------------------------------------------------------------
// LayerNorm over D=768; one block (256 threads) per row.
// mode 0: bf16 out, no pos        mode 1: bf16 out + pos[row%196]
// mode 2: f32 out                 mode 3: bf16 out, permute (b,t,n)->(b,n,t), + pos[t]
// ---------------------------------------------------------------------------
__global__ __launch_bounds__(256)
void ln_kernel(const float* __restrict__ x, const float* __restrict__ g,
               const float* __restrict__ bb, bf16* __restrict__ outB,
               float* __restrict__ outF, const float* __restrict__ pos, int mode) {
  int row = blockIdx.x, tid = threadIdx.x;
  const float* xr = x + (size_t)row * DD;
  float v0 = xr[tid], v1 = xr[tid + 256], v2 = xr[tid + 512];
  __shared__ float r1[256], r2[256];
  r1[tid] = v0 + v1 + v2;
  r2[tid] = v0 * v0 + v1 * v1 + v2 * v2;
  __syncthreads();
  for (int st = 128; st > 0; st >>= 1) {
    if (tid < st) { r1[tid] += r1[tid + st]; r2[tid] += r2[tid + st]; }
    __syncthreads();
  }
  float mean = r1[0] * (1.0f / DD);
  float var  = r2[0] * (1.0f / DD) - mean * mean;
  float rstd = rsqrtf(var + 1e-6f);

  size_t orow = row;
  const float* prow = nullptr;
  if (mode == 1) {
    prow = pos + (size_t)(row % NN) * DD;
  } else if (mode == 3) {
    int b = row / (TT * NN), rem = row % (TT * NN);
    int t = rem / NN, n = rem % NN;
    orow = (size_t)((b * NN + n) * TT + t);
    prow = pos + (size_t)t * DD;
  }
  float vv[3] = {v0, v1, v2};
#pragma unroll
  for (int j = 0; j < 3; ++j) {
    int c = tid + j * 256;
    float y = (vv[j] - mean) * rstd * g[c] + bb[c];
    if (prow) y += prow[c];
    if (mode == 2) outF[orow * DD + c] = y;
    else          outB[orow * DD + c] = (bf16)y;
  }
}

// ---------------------------------------------------------------------------
// s2t kv input: row bt*196+n = sn[bt*197+1+n] + clip_space_pos[n]   (bf16)
// ---------------------------------------------------------------------------
__global__ void build_spat_kernel(const float* __restrict__ sn, const float* __restrict__ pos,
                                  bf16* __restrict__ out) {
  int idx = blockIdx.x * 256 + threadIdx.x;  // MT*DD total
  int r = idx / DD, c = idx - r * DD;
  int bt = r / NN, n = r - bt * NN;
  out[idx] = (bf16)(sn[((size_t)(bt * 197 + 1 + n)) * DD + c] + pos[(size_t)n * DD + c]);
}

// t2s q input: row (b*196+n)*8+t = sn[(b*8+t)*197+1+n] + clip_time_pos[t]  (bf16)
__global__ void build_sq_kernel(const float* __restrict__ sn, const float* __restrict__ pos,
                                bf16* __restrict__ out) {
  int idx = blockIdx.x * 256 + threadIdx.x;  // MT*DD total
  int r = idx / DD, c = idx - r * DD;
  int b = r / (NN * TT), rem = r - b * (NN * TT);
  int n = rem / TT, t = rem - n * TT;
  size_t src = (size_t)((b * TT + t) * 197 + 1 + n);
  out[idx] = (bf16)(sn[src * DD + c] + pos[(size_t)t * DD + c]);
}

// cls rows of t2s_full <- sn cls rows (f32 + bf16)
__global__ void copy_cls_kernel(const float* __restrict__ sn, float* __restrict__ of,
                                bf16* __restrict__ ob) {
  int idx = blockIdx.x * 256 + threadIdx.x;  // 32*DD total
  int b = idx / DD, c = idx - b * DD;
  size_t o = (size_t)(b * 197) * DD + c;
  float v = sn[o];
  of[o] = v;
  ob[o] = (bf16)v;
}

// ---------------------------------------------------------------------------
// GEMM: C[M,N] = act(A_bf16[M,K] @ W_bf16[N,K]^T + bias) (+res0 +res1)
// Block: 128 threads (4 waves); wave computes 16x64 of a 64x64 block tile.
// B tile (64 N-rows x 32 K) is staged once per block into double-buffered LDS
// via gfx1250 async global->LDS DMA (ASYNCcnt), overlapping next-tile DMA with
// current-tile WMMAs. Falls back to load+ds_store if builtins unavailable.
// rowmap 1: t2s permute (b,n,t)->s-layout row (b*8+t)*197+1+n
// ---------------------------------------------------------------------------
__global__ __launch_bounds__(128)
void gemm_kernel(const bf16* __restrict__ A, int lda,
                 const bf16* __restrict__ W, int ldw,
                 const float* __restrict__ bias,
                 float* __restrict__ outF, bf16* __restrict__ outB, int ldo,
                 const float* __restrict__ res0, const float* __restrict__ res1,
                 int M, int N, int K, int doGelu, int rowmap) {
  __shared__ bf16 Bs[2][64][32];  // 8 KB double-buffered B tile

  const int tid  = threadIdx.x;
  const int lane = tid & 31;
  const int wave = tid >> 5;
  const int col  = lane & 15;
  const int hi   = lane >> 4;
  const int m0 = blockIdx.y * 64 + wave * 16;
  const int n0 = blockIdx.x * 64;

  int arow = m0 + col; if (arow >= M) arow = M - 1;
  const bf16* Ap = A + (size_t)arow * lda + hi * 16;

  // B-tile staging: thread tid covers row n = tid>>1, half = tid&1 (16 elems)
  const int brow = tid >> 1, bhalf = tid & 1;
  const bf16* Bg = W + (size_t)(n0 + brow) * ldw + bhalf * 16;
  bf16* Bl0 = &Bs[0][brow][bhalf * 16];
  bf16* Bl1 = &Bs[1][brow][bhalf * 16];

  auto stage = [&](int buf, int k0) {
    const bf16* g = Bg + k0;
    bf16* l = buf ? Bl1 : Bl0;
#ifdef USE_ASYNC_LDS
    __builtin_amdgcn_global_load_async_to_lds_b128((g_i32x4*)g, (l_i32x4*)l, 0, 0);
    __builtin_amdgcn_global_load_async_to_lds_b128((g_i32x4*)(g + 8), (l_i32x4*)(l + 8), 0, 0);
#else
    bf16x8 r0 = *(const bf16x8*)(g);
    bf16x8 r1 = *(const bf16x8*)(g + 8);
    *(bf16x8*)(l) = r0;
    *(bf16x8*)(l + 8) = r1;
#endif
  };

  v8f acc[4];
#pragma unroll
  for (int t = 0; t < 4; ++t)
#pragma unroll
    for (int i = 0; i < 8; ++i) acc[t][i] = 0.0f;

  const int nk = K >> 5;  // K multiple of 32
  stage(0, 0);
  for (int i = 0; i < nk; ++i) {
    const int cur = i & 1;
#ifdef USE_ASYNC_LDS
    __builtin_amdgcn_s_wait_asynccnt(0);
#endif
    __syncthreads();  // B[cur] visible; prior reads of B[cur^1] all consumed

    // A fragment (per-wave, distinct rows) + B fragments from LDS
    v16bf a = pack2(*(const bf16x8*)(Ap + i * 32),
                    *(const bf16x8*)(Ap + i * 32 + 8));
    v16bf bfrag[4];
#pragma unroll
    for (int t = 0; t < 4; ++t) {
      const bf16* l = &Bs[cur][t * 16 + col][hi * 16];
      bfrag[t] = pack2(*(const bf16x8*)(l), *(const bf16x8*)(l + 8));
    }
    // Kick off next tile's DMA into the other buffer (overlaps the WMMAs).
    if (i + 1 < nk) stage(cur ^ 1, (i + 1) * 32);
#pragma unroll
    for (int t = 0; t < 4; ++t) acc[t] = wmma_bf16(a, bfrag[t], acc[t]);
  }

#pragma unroll
  for (int t = 0; t < 4; ++t) {
    int c = n0 + t * 16 + col;
    float bv = bias ? bias[c] : 0.0f;
#pragma unroll
    for (int i = 0; i < 8; ++i) {
      int row = m0 + i + hi * 8;
      if (row < M) {
        float v = acc[t][i] + bv;
        if (doGelu) v = gelu_exact(v);
        int orow = row;
        if (rowmap == 1) {
          int b = row / (NN * TT), rem = row % (NN * TT);
          int n = rem / TT, tt = rem % TT;
          orow = (b * TT + tt) * 197 + 1 + n;
        }
        size_t o = (size_t)orow * ldo + c;
        if (res0) v += res0[o];
        if (res1) v += res1[o];
        if (outF) outF[o] = v;
        if (outB) outB[o] = (bf16)v;
      }
    }
  }
}

// ---------------------------------------------------------------------------
// Flash attention (1 wave / block). grid = (ceil(Lq/16), HEADS, B)
// Q/K/V bf16, head h at column h*64 of its buffer; scores scaled; keys >= Lk
// masked to -inf. Output bf16, row-major ldo, head offset h*64.
// ---------------------------------------------------------------------------
__global__ __launch_bounds__(32)
void attn_kernel(const bf16* __restrict__ Q, int ldq,
                 const bf16* __restrict__ Kp, int ldk,
                 const bf16* __restrict__ Vp, int ldv,
                 bf16* __restrict__ O, int ldo,
                 int Lq, int Lk, float scale) {
  __shared__ bf16 plds[16][32];
  const int lane = threadIdx.x;
  const int col = lane & 15;
  const int hi  = lane >> 4;
  const int qt = blockIdx.x, h = blockIdx.y, b = blockIdx.z;
  const size_t qbase = (size_t)b * Lq;
  const size_t kbase = (size_t)b * Lk;
  const int hc = h * HD;

  // Q A-fragments (d 0..31 and 32..63)
  int qrow = qt * 16 + col; if (qrow >= Lq) qrow = Lq - 1;
  const bf16* qp = Q + (qbase + qrow) * (size_t)ldq + hc + hi * 16;
  v16bf a0 = pack2(*(const bf16x8*)(qp),      *(const bf16x8*)(qp + 8));
  v16bf a1 = pack2(*(const bf16x8*)(qp + 32), *(const bf16x8*)(qp + 40));

  float rmax[8], rsum[8];
  v8f ov[4];
#pragma unroll
  for (int i = 0; i < 8; ++i) { rmax[i] = -__builtin_inff(); rsum[i] = 0.0f; }
#pragma unroll
  for (int t = 0; t < 4; ++t)
#pragma unroll
    for (int i = 0; i < 8; ++i) ov[t][i] = 0.0f;

  for (int j0 = 0; j0 < Lk; j0 += 32) {
    v8f sv[2];
#pragma unroll
    for (int kk = 0; kk < 2; ++kk) {
      int key = j0 + kk * 16 + col; if (key >= Lk) key = Lk - 1;
      const bf16* kp = Kp + (kbase + key) * (size_t)ldk + hc + hi * 16;
      v16bf b0 = pack2(*(const bf16x8*)(kp),      *(const bf16x8*)(kp + 8));
      v16bf b1 = pack2(*(const bf16x8*)(kp + 32), *(const bf16x8*)(kp + 40));
      v8f z;
#pragma unroll
      for (int i = 0; i < 8; ++i) z[i] = 0.0f;
      z = wmma_bf16(a0, b0, z);
      z = wmma_bf16(a1, b1, z);
      sv[kk] = z;
    }
    // online softmax over these 32 keys
#pragma unroll
    for (int i = 0; i < 8; ++i) {
      float x0 = (j0 + col      < Lk) ? sv[0][i] * scale : -__builtin_inff();
      float x1 = (j0 + 16 + col < Lk) ? sv[1][i] * scale : -__builtin_inff();
      float tm = fmaxf(x0, x1);
      tm = fmaxf(tm, __shfl_xor(tm, 1, 32));
      tm = fmaxf(tm, __shfl_xor(tm, 2, 32));
      tm = fmaxf(tm, __shfl_xor(tm, 4, 32));
      tm = fmaxf(tm, __shfl_xor(tm, 8, 32));
      float nm = fmaxf(rmax[i], tm);
      float al = __expf(rmax[i] - nm);
      float p0 = __expf(x0 - nm);
      float p1 = __expf(x1 - nm);
      float ps = p0 + p1;
      ps += __shfl_xor(ps, 1, 32);
      ps += __shfl_xor(ps, 2, 32);
      ps += __shfl_xor(ps, 4, 32);
      ps += __shfl_xor(ps, 8, 32);
      rsum[i] = rsum[i] * al + ps;
      rmax[i] = nm;
      ov[0][i] *= al; ov[1][i] *= al; ov[2][i] *= al; ov[3][i] *= al;
      plds[i + hi * 8][col]      = (bf16)p0;
      plds[i + hi * 8][col + 16] = (bf16)p1;
    }
    __syncthreads();
    // P as A-fragment (16 q-rows x 32 keys)
    const bf16* pp = &plds[col][hi * 8];
    v16bf pa = pack2(*(const bf16x8*)(pp), *(const bf16x8*)(pp + 16));
    // V B-fragments, O += P*V
#pragma unroll
    for (int t = 0; t < 4; ++t) {
      v16bf vb;
#pragma unroll
      for (int e = 0; e < 16; ++e) {
        int key = j0 + hi * 8 + (e < 8 ? e : e + 8); if (key >= Lk) key = Lk - 1;  // p==0 there
        vb[e] = Vp[(kbase + key) * (size_t)ldv + hc + t * 16 + col];
      }
      ov[t] = wmma_bf16(pa, vb, ov[t]);
    }
    __syncthreads();
  }

#pragma unroll
  for (int i = 0; i < 8; ++i) {
    int r = qt * 16 + i + hi * 8;
    if (r < Lq) {
      float inv = 1.0f / rsum[i];
      size_t ro = (qbase + r) * (size_t)ldo + hc;
      O[ro + 0 * 16 + col] = (bf16)(ov[0][i] * inv);
      O[ro + 1 * 16 + col] = (bf16)(ov[1][i] * inv);
      O[ro + 2 * 16 + col] = (bf16)(ov[2][i] * inv);
      O[ro + 3 * 16 + col] = (bf16)(ov[3][i] * inv);
    }
  }
}

// ---------------------------------------------------------------------------
// Host orchestration
// ---------------------------------------------------------------------------
static inline void cvt(hipStream_t s, const float* in, bf16* out, int n) {
  f2b_kernel<<<(n + 255) / 256, 256, 0, s>>>(in, out, n);
}
static inline void gemm(hipStream_t s, const bf16* A, int lda, const bf16* W, int ldw,
                        const float* bias, float* oF, bf16* oB, int ldo,
                        const float* r0, const float* r1, int M, int N, int K,
                        int gelu, int rowmap) {
  dim3 g(N / 64, (M + 63) / 64);
  gemm_kernel<<<g, 128, 0, s>>>(A, lda, W, ldw, bias, oF, oB, ldo, r0, r1, M, N, K, gelu, rowmap);
}

extern "C" void kernel_launch(void* const* d_in, const int* in_sizes, int n_in,
                              void* d_out, int out_size, void* d_ws, size_t ws_size,
                              hipStream_t stream) {
  (void)in_sizes; (void)n_in; (void)out_size; (void)ws_size;
  const float* s_x  = (const float*)d_in[0];
  const float* t_x  = (const float*)d_in[1];
  const float* ln1_g = (const float*)d_in[2];
  const float* ln1_b = (const float*)d_in[3];
  const float* qkv_w = (const float*)d_in[4];
  const float* q_bias = (const float*)d_in[5];
  const float* v_bias = (const float*)d_in[6];
  const float* attn_proj_w = (const float*)d_in[7];
  const float* attn_proj_b = (const float*)d_in[8];
  const float* lns_g = (const float*)d_in[9];
  const float* lns_b = (const float*)d_in[10];
  const float* lnt_g = (const float*)d_in[11];
  const float* lnt_b = (const float*)d_in[12];
  const float* clip_space_pos = (const float*)d_in[13];
  const float* vmae_space_pos = (const float*)d_in[14];
  const float* s2t_q_w = (const float*)d_in[15];
  const float* s2t_q_b = (const float*)d_in[16];
  const float* s2t_kv_w = (const float*)d_in[17];
  const float* s2t_kv_b = (const float*)d_in[18];
  const float* s2t_proj_w = (const float*)d_in[19];
  const float* s2t_proj_b = (const float*)d_in[20];
  const float* s2t_f1_w = (const float*)d_in[21];
  const float* s2t_f1_b = (const float*)d_in[22];
  const float* s2t_f2_w = (const float*)d_in[23];
  const float* s2t_f2_b = (const float*)d_in[24];
  const float* clip_time_pos = (const float*)d_in[25];
  const float* vmae_time_pos = (const float*)d_in[26];
  const float* t2s_q_w = (const float*)d_in[27];
  const float* t2s_q_b = (const float*)d_in[28];
  const float* t2s_kv_w = (const float*)d_in[29];
  const float* t2s_kv_b = (const float*)d_in[30];
  const float* t2s_proj_w = (const float*)d_in[31];
  const float* t2s_proj_b = (const float*)d_in[32];
  const float* t2s_f1_w = (const float*)d_in[33];
  const float* t2s_f1_b = (const float*)d_in[34];
  const float* t2s_f2_w = (const float*)d_in[35];
  const float* t2s_f2_b = (const float*)d_in[36];
  const float* ln2_g = (const float*)d_in[37];
  const float* ln2_b = (const float*)d_in[38];
  const float* mlp1_w = (const float*)d_in[39];
  const float* mlp1_b = (const float*)d_in[40];
  const float* mlp2_w = (const float*)d_in[41];
  const float* mlp2_b = (const float*)d_in[42];
  const float* ln2s_g = (const float*)d_in[43];
  const float* ln2s_b = (const float*)d_in[44];
  const float* smlp1_w = (const float*)d_in[45];
  const float* smlp1_b = (const float*)d_in[46];
  const float* smlp2_w = (const float*)d_in[47];
  const float* smlp2_b = (const float*)d_in[48];

  float* outS = (float*)d_out;                       // s: 32*197*768
  float* outT = outS + (size_t)MS * DD;              // t: 4*1568*768

  // ---- workspace arena (deterministic layout) ----
  char* base = (char*)d_ws;
  size_t off = 0;
  auto take = [&](size_t bytes) { size_t o = off; off += (bytes + 255) & ~(size_t)255; return o; };
  auto BF = [](size_t n) { return n * sizeof(bf16); };
  auto F4 = [](size_t n) { return n * sizeof(float); };

  size_t o_wqkv  = take(BF((size_t)3 * DD * DD));
  size_t o_wproj = take(BF((size_t)DD * DD));
  size_t o_ws2tq = take(BF((size_t)DD * DD));
  size_t o_ws2tkv = take(BF((size_t)2 * DD * DD));
  size_t o_ws2tp = take(BF((size_t)DD * DD));
  size_t o_ws2tf1 = take(BF((size_t)DOWN * DD));
  size_t o_ws2tf2 = take(BF((size_t)DD * DOWN));
  size_t o_wt2sq = take(BF((size_t)DD * DD));
  size_t o_wt2skv = take(BF((size_t)2 * DD * DD));
  size_t o_wt2sp = take(BF((size_t)DD * DD));
  size_t o_wt2sf1 = take(BF((size_t)DOWN * DD));
  size_t o_wt2sf2 = take(BF((size_t)DD * DOWN));
  size_t o_wm1 = take(BF((size_t)HID * DD));
  size_t o_wm2 = take(BF((size_t)DD * HID));
  size_t o_wsm1 = take(BF((size_t)HID * DD));
  size_t o_wsm2 = take(BF((size_t)DD * HID));
  size_t o_qkvb = take(F4(3 * DD));
  size_t o_sn   = take(F4((size_t)MS * DD));   // LN(s_x) f32
  size_t o_tbuf = take(F4((size_t)MT * DD));   // t after self-attn
  size_t o_s2to = take(F4((size_t)MT * DD));   // s2t proj out f32
  size_t o_t2sf = take(F4((size_t)MS * DD));   // t2s full f32
  size_t o_pool = take((size_t)72 * 1024 * 1024);

  bf16* wqkv = (bf16*)(base + o_wqkv);
  bf16* wproj = (bf16*)(base + o_wproj);
  bf16* ws2tq = (bf16*)(base + o_ws2tq);
  bf16* ws2tkv = (bf16*)(base + o_ws2tkv);
  bf16* ws2tp = (bf16*)(base + o_ws2tp);
  bf16* ws2tf1 = (bf16*)(base + o_ws2tf1);
  bf16* ws2tf2 = (bf16*)(base + o_ws2tf2);
  bf16* wt2sq = (bf16*)(base + o_wt2sq);
  bf16* wt2skv = (bf16*)(base + o_wt2skv);
  bf16* wt2sp = (bf16*)(base + o_wt2sp);
  bf16* wt2sf1 = (bf16*)(base + o_wt2sf1);
  bf16* wt2sf2 = (bf16*)(base + o_wt2sf2);
  bf16* wm1 = (bf16*)(base + o_wm1);
  bf16* wm2 = (bf16*)(base + o_wm2);
  bf16* wsm1 = (bf16*)(base + o_wsm1);
  bf16* wsm2 = (bf16*)(base + o_wsm2);
  float* qkvb = (float*)(base + o_qkvb);
  float* sn_f = (float*)(base + o_sn);
  float* tbuf = (float*)(base + o_tbuf);
  float* s2to = (float*)(base + o_s2to);
  float* t2sf = (float*)(base + o_t2sf);
  char* pool = base + o_pool;

  const float scale = 0.125f;  // HD^-0.5 = 64^-0.5

  // Phase 0: weights -> bf16 (L2-resident working set), fused qkv bias
  cvt(stream, qkv_w, wqkv, 3 * DD * DD);
  cvt(stream, attn_proj_w, wproj, DD * DD);
  cvt(stream, s2t_q_w, ws2tq, DD * DD);
  cvt(stream, s2t_kv_w, ws2tkv, 2 * DD * DD);
  cvt(stream, s2t_proj_w, ws2tp, DD * DD);
  cvt(stream, s2t_f1_w, ws2tf1, DOWN * DD);
  cvt(stream, s2t_f2_w, ws2tf2, DD * DOWN);
  cvt(stream, t2s_q_w, wt2sq, DD * DD);
  cvt(stream, t2s_kv_w, wt2skv, 2 * DD * DD);
  cvt(stream, t2s_proj_w, wt2sp, DD * DD);
  cvt(stream, t2s_f1_w, wt2sf1, DOWN * DD);
  cvt(stream, t2s_f2_w, wt2sf2, DD * DOWN);
  cvt(stream, mlp1_w, wm1, HID * DD);
  cvt(stream, mlp2_w, wm2, DD * HID);
  cvt(stream, smlp1_w, wsm1, HID * DD);
  cvt(stream, smlp2_w, wsm2, DD * HID);
  qkv_bias_kernel<<<(3 * DD + 255) / 256, 256, 0, stream>>>(q_bias, v_bias, qkvb);

  // ---- Phase 1: t self-attention ----
  bf16* p_tn  = (bf16*)(pool);
  bf16* p_qkv = (bf16*)(pool + BF((size_t)MT * DD) + 256);
  bf16* p_ao  = (bf16*)(pool + BF((size_t)MT * DD) + BF((size_t)MT * 3 * DD) + 512);

  ln_kernel<<<MT, 256, 0, stream>>>(t_x, ln1_g, ln1_b, p_tn, nullptr, nullptr, 0);
  gemm(stream, p_tn, DD, wqkv, DD, qkvb, nullptr, p_qkv, 3 * DD, nullptr, nullptr,
       MT, 3 * DD, DD, 0, 0);
  {
    dim3 g((TT * NN + 15) / 16, HEADS, 4);
    attn_kernel<<<g, 32, 0, stream>>>(p_qkv, 3 * DD, p_qkv + DD, 3 * DD, p_qkv + 2 * DD, 3 * DD,
                                      p_ao, DD, TT * NN, TT * NN, scale);
  }
  gemm(stream, p_ao, DD, wproj, DD, attn_proj_b, tbuf, nullptr, DD, t_x, nullptr,
       MT, DD, DD, 0, 0);

  // ---- Phase 2: sn + s2t cross-attention + adapter ----
  ln_kernel<<<MS, 256, 0, stream>>>(s_x, lns_g, lns_b, nullptr, sn_f, nullptr, 2);

  bf16* p_tq   = (bf16*)(pool);
  bf16* p_spat = (bf16*)((char*)p_tq + BF((size_t)MT * DD) + 256);
  bf16* p_q1   = (bf16*)((char*)p_spat + BF((size_t)MT * DD) + 256);
  bf16* p_kv1  = (bf16*)((char*)p_q1 + BF((size_t)MT * DD) + 256);
  bf16* p_x1   = (bf16*)((char*)p_kv1 + BF((size_t)MT * 2 * DD) + 256);
  bf16* p_s2tb = (bf16*)((char*)p_x1 + BF((size_t)MT * DD) + 256);
  bf16* p_h1   = (bf16*)((char*)p_s2tb + BF((size_t)MT * DD) + 256);

  ln_kernel<<<MT, 256, 0, stream>>>(tbuf, lnt_g, lnt_b, p_tq, nullptr, vmae_space_pos, 1);
  build_spat_kernel<<<(MT * DD) / 256, 256, 0, stream>>>(sn_f, clip_space_pos, p_spat);
  gemm(stream, p_tq, DD, ws2tq, DD, s2t_q_b, nullptr, p_q1, DD, nullptr, nullptr, MT, DD, DD, 0, 0);
  gemm(stream, p_spat, DD, ws2tkv, DD, s2t_kv_b, nullptr, p_kv1, 2 * DD, nullptr, nullptr,
       MT, 2 * DD, DD, 0, 0);
  {
    dim3 g((NN + 15) / 16, HEADS, 32);
    attn_kernel<<<g, 32, 0, stream>>>(p_q1, DD, p_kv1, 2 * DD, p_kv1 + DD, 2 * DD,
                                      p_x1, DD, NN, NN, scale);
  }
  gemm(stream, p_x1, DD, ws2tp, DD, s2t_proj_b, s2to, p_s2tb, DD, nullptr, nullptr,
       MT, DD, DD, 0, 0);
  gemm(stream, p_s2tb, DD, ws2tf1, DD, s2t_f1_b, nullptr, p_h1, DOWN, nullptr, nullptr,
       MT, DOWN, DD, 1, 0);
  // t = t + adapter(s2t): outT = h1@f2^T + b + s2t_out + tbuf
  gemm(stream, p_h1, DOWN, ws2tf2, DOWN, s2t_f2_b, outT, nullptr, DD, s2to, tbuf,
       MT, DD, DOWN, 0, 0);

  // ---- Phase 3: t2s cross-attention + adapter ----
  bf16* p_sq   = (bf16*)(pool);
  bf16* p_kvin = (bf16*)((char*)p_sq + BF((size_t)MT * DD) + 256);
  bf16* p_q2   = (bf16*)((char*)p_kvin + BF((size_t)MT * DD) + 256);
  bf16* p_kv2  = (bf16*)((char*)p_q2 + BF((size_t)MT * DD) + 256);
  bf16* p_x2   = (bf16*)((char*)p_kv2 + BF((size_t)MT * 2 * DD) + 256);
  bf16* p_t2sb = (bf16*)((char*)p_x2 + BF((size_t)MT * DD) + 256);
  bf16* p_h2   = (bf16*)((char*)p_t2sb + BF((size_t)MS * DD) + 256);

  ln_kernel<<<MT, 256, 0, stream>>>(outT, lnt_g, lnt_b, p_kvin, nullptr, vmae_time_pos, 3);
  build_sq_kernel<<<(MT * DD) / 256, 256, 0, stream>>>(sn_f, clip_time_pos, p_sq);
  gemm(stream, p_sq, DD, wt2sq, DD, t2s_q_b, nullptr, p_q2, DD, nullptr, nullptr, MT, DD, DD, 0, 0);
  gemm(stream, p_kvin, DD, wt2skv, DD, t2s_kv_b, nullptr, p_kv2, 2 * DD, nullptr, nullptr,
       MT, 2 * DD, DD, 0, 0);
  {
    dim3 g(1, HEADS, 4 * NN);  // Lq = Lk = 8, 784 (b,n) batches
    attn_kernel<<<g, 32, 0, stream>>>(p_q2, DD, p_kv2, 2 * DD, p_kv2 + DD, 2 * DD,
                                      p_x2, DD, TT, TT, scale);
  }
  // proj writes permuted into s-layout rows (patch rows); cls rows copied from sn
  gemm(stream, p_x2, DD, wt2sp, DD, t2s_proj_b, t2sf, p_t2sb, DD, nullptr, nullptr,
       MT, DD, DD, 0, 1);
  copy_cls_kernel<<<(32 * DD) / 256, 256, 0, stream>>>(sn_f, t2sf, p_t2sb);
  gemm(stream, p_t2sb, DD, wt2sf1, DD, t2s_f1_b, nullptr, p_h2, DOWN, nullptr, nullptr,
       MS, DOWN, DD, 1, 0);
  // s = s_x + adapter(t2s): outS = h2@f2^T + b + t2s_full + s_x
  gemm(stream, p_h2, DOWN, wt2sf2, DOWN, t2s_f2_b, outS, nullptr, DD, t2sf, s_x,
       MS, DD, DOWN, 0, 0);

  // ---- Phase 4: t MLP ----
  bf16* p_tn3  = (bf16*)(pool);
  bf16* p_hmlp = (bf16*)((char*)p_tn3 + BF((size_t)MT * DD) + 256);
  ln_kernel<<<MT, 256, 0, stream>>>(outT, ln2_g, ln2_b, p_tn3, nullptr, nullptr, 0);
  gemm(stream, p_tn3, DD, wm1, DD, mlp1_b, nullptr, p_hmlp, HID, nullptr, nullptr,
       MT, HID, DD, 1, 0);
  gemm(stream, p_hmlp, HID, wm2, HID, mlp2_b, outT, nullptr, DD, outT, nullptr,
       MT, DD, HID, 0, 0);

  // ---- Phase 5: s MLP ----
  bf16* p_sn2 = (bf16*)(pool);
  bf16* p_hs  = (bf16*)((char*)p_sn2 + BF((size_t)MS * DD) + 256);
  ln_kernel<<<MS, 256, 0, stream>>>(outS, ln2s_g, ln2s_b, p_sn2, nullptr, nullptr, 0);
  gemm(stream, p_sn2, DD, wsm1, DD, smlp1_b, nullptr, p_hs, HID, nullptr, nullptr,
       MS, HID, DD, 1, 0);
  gemm(stream, p_hs, HID, wsm2, HID, smlp2_b, outS, nullptr, DD, outS, nullptr,
       MS, DD, HID, 0, 0);
}
